// AutoSparseLinear_64965675319396
// MI455X (gfx1250) — matrix operand order repaired and checked by reading, not compiled, not run on hardware
//
#include <hip/hip_runtime.h>

// ---- problem constants (from the reference) ----
#define BATCH   512
#define H_IN    128
#define W_IN    64
#define H_OUT   256
#define W_OUT   64
#define KGATH   4
#define DDIM    256          // W_IN * KGATH

// ---- tiling ----
#define BM      64           // batch rows per workgroup
#define THREADS 128          // 4 wave32s
#define LDSTR   264          // padded LDS row stride in bf16 elems (conflict-free)

typedef __bf16 bf16_t;
typedef __attribute__((ext_vector_type(16))) __bf16 v16bf;
typedef __attribute__((ext_vector_type(8)))  __bf16 v8bf;
typedef __attribute__((ext_vector_type(8)))  float  v8f;

__global__ __launch_bounds__(THREADS)
void sparse_linear_wmma(const float* __restrict__ x,
                        const long long* __restrict__ mask,
                        const float* __restrict__ Wmat,
                        const float* __restrict__ bias,
                        float* __restrict__ out)
{
    extern __shared__ char smem[];
    bf16_t* As = reinterpret_cast<bf16_t*>(smem);                       // [BM][LDSTR]
    bf16_t* Bs = reinterpret_cast<bf16_t*>(smem + (size_t)BM * LDSTR * 2); // [W_OUT][LDSTR]

    const int h   = blockIdx.y;
    const int b0  = blockIdx.x * BM;
    const int tid = threadIdx.x;

    // gather indices for this head (uniform across the block)
    int midx[KGATH];
    #pragma unroll
    for (int j = 0; j < KGATH; ++j) midx[j] = (int)mask[h * KGATH + j];

    // ---- stage A: gathered x rows, f32 -> bf16, [BM][DDIM] ----
    for (int q = tid; q < BM * (DDIM / 4); q += THREADS) {
        const int row = q >> 6;            // DDIM/4 = 64 float4 chunks per row
        const int d   = (q & 63) * 4;
        const int j   = d >> 6;            // which gathered chunk
        const int dj  = d & 63;            // offset inside the x row
        const float4 v = *reinterpret_cast<const float4*>(
            x + ((size_t)(b0 + row) * H_IN + midx[j]) * W_IN + dj);
        bf16_t* dst = As + row * LDSTR + d;
        dst[0] = (bf16_t)v.x; dst[1] = (bf16_t)v.y;
        dst[2] = (bf16_t)v.z; dst[3] = (bf16_t)v.w;
    }

    // ---- stage B: W_h, f32 -> bf16, [W_OUT][DDIM] (already K-contiguous) ----
    const float* Wh = Wmat + (size_t)h * W_OUT * DDIM;
    for (int q = tid; q < W_OUT * (DDIM / 4); q += THREADS) {
        const int n = q >> 6;
        const int d = (q & 63) * 4;
        const float4 v = *reinterpret_cast<const float4*>(Wh + (size_t)n * DDIM + d);
        bf16_t* dst = Bs + n * LDSTR + d;
        dst[0] = (bf16_t)v.x; dst[1] = (bf16_t)v.y;
        dst[2] = (bf16_t)v.z; dst[3] = (bf16_t)v.w;
    }

    __syncthreads();

    // ---- per-wave 16x64 output strip, 4 WMMA n-tiles ----
    const int wave = tid >> 5;
    const int lane = tid & 31;
    const int half = lane >> 4;     // 0: lanes 0-15, 1: lanes 16-31
    const int ln   = lane & 15;
    const int arow = wave * 16 + ln;

    v8f acc[4];
    #pragma unroll
    for (int nt = 0; nt < 4; ++nt) {
        const float bv = bias[h * W_OUT + nt * 16 + ln];   // per-column bias
        #pragma unroll
        for (int i = 0; i < 8; ++i) acc[nt][i] = bv;       // fused bias init
    }

    for (int k0 = 0; k0 < DDIM; k0 += 32) {
        // A fragment: lane holds row arow; elems 0-7  = K k0+8*half+{0..7}
        //                                  elems 8-15 = K k0+16+8*half+{0..7}
        const bf16_t* ap = As + arow * LDSTR + k0 + 8 * half;
        const v8bf a_lo = *reinterpret_cast<const v8bf*>(ap);
        const v8bf a_hi = *reinterpret_cast<const v8bf*>(ap + 16);
        v16bf afrag;
        #pragma unroll
        for (int i = 0; i < 8; ++i) { afrag[i] = a_lo[i]; afrag[i + 8] = a_hi[i]; }

        #pragma unroll
        for (int nt = 0; nt < 4; ++nt) {
            // B fragment: lane holds column nt*16+ln; elems 0-7 = K k0+8*half+{0..7}, elems 8-15 = K k0+16+8*half+{0..7}
            const bf16_t* bp = Bs + (nt * 16 + ln) * LDSTR + k0 + 8 * half;
            const v8bf b_lo = *reinterpret_cast<const v8bf*>(bp);
            const v8bf b_hi = *reinterpret_cast<const v8bf*>(bp + 16);
            v16bf bfrag;
            #pragma unroll
            for (int i = 0; i < 8; ++i) { bfrag[i] = b_lo[i]; bfrag[i + 8] = b_hi[i]; }

            acc[nt] = __builtin_amdgcn_wmma_f32_16x16x32_bf16(
                false, afrag, false, bfrag, (short)0, acc[nt], false, false);
        }
    }

    // ---- store: C/D layout -> out[b][h][o], VGPR v: row = v + 8*half, col = ln ----
    #pragma unroll
    for (int nt = 0; nt < 4; ++nt) {
        const int o = nt * 16 + ln;
        #pragma unroll
        for (int v = 0; v < 8; ++v) {
            const int brow = b0 + wave * 16 + v + 8 * half;
            out[((size_t)brow * H_OUT + h) * W_OUT + o] = acc[nt][v];
        }
    }
}

extern "C" void kernel_launch(void* const* d_in, const int* in_sizes, int n_in,
                              void* d_out, int out_size, void* d_ws, size_t ws_size,
                              hipStream_t stream) {
    const float*     x    = (const float*)d_in[0];
    const long long* mask = (const long long*)d_in[1];
    const float*     Wm   = (const float*)d_in[2];
    const float*     b    = (const float*)d_in[3];
    float*           out  = (float*)d_out;

    dim3 grid(BATCH / BM, H_OUT, 1);
    const size_t lds_bytes = (size_t)(BM + W_OUT) * LDSTR * sizeof(bf16_t); // 67584 B
    sparse_linear_wmma<<<grid, THREADS, lds_bytes, stream>>>(x, mask, Wm, b, out);
}
